// WindowAttention_60997125537800
// MI455X (gfx1250) — matrix operand order repaired and checked by reading, not compiled, not run on hardware
//
#include <hip/hip_runtime.h>
#include <math.h>

// ---------------- problem constants ----------------
#define WB   2048          // total windows (batch * NW)
#define NW   64            // windows per image
#define S    49            // tokens per window
#define SP   64            // padded tokens
#define C    512           // embedding
#define H    16            // heads
#define D    32            // head dim
#define N3   1536          // 3*C
#define MROWS (WB * S)     // 100352 GEMM rows, divisible by 64

typedef __bf16 bf16_t;
typedef __attribute__((ext_vector_type(16))) __bf16 v16bf;
typedef __attribute__((ext_vector_type(8)))  __bf16 v8bf;
typedef __attribute__((ext_vector_type(8)))  float  v8f;
typedef __attribute__((ext_vector_type(4)))  float  v4f;
typedef __attribute__((ext_vector_type(4)))  uint32_t u32x4;
typedef __attribute__((ext_vector_type(8)))  uint32_t u32x8;

static __device__ __forceinline__ v16bf cat16(v8bf lo, v8bf hi) {
    return __builtin_shufflevector(lo, hi, 0,1,2,3,4,5,6,7,8,9,10,11,12,13,14,15);
}
static __device__ __forceinline__ v8f zero8() {
    v8f z = {0.f,0.f,0.f,0.f,0.f,0.f,0.f,0.f};
    return z;
}
static __device__ __forceinline__ v8f wmma_bf16(v16bf a, v16bf b, v8f c) {
    // D(f32 16x16) = A(bf16 16x32) * B(bf16 32x16) + C
    return __builtin_amdgcn_wmma_f32_16x16x32_bf16(false, a, false, b,
                                                   (short)0, c, false, false);
}

// A-fragment from row-major bf16 LDS tile:
// lane l<16: row m=l, K={0..7,16..23}; lane>=16: K={8..15,24..31}.
static __device__ __forceinline__ v16bf load_afrag(const bf16_t* rowPtr, int hl) {
    v8bf lo = *(const v8bf*)(rowPtr + hl * 8);
    v8bf hi = *(const v8bf*)(rowPtr + 16 + hl * 8);
    return cat16(lo, hi);
}
// B-fragment from N-major bf16 LDS tile (K contiguous per column):
// lane l<16: column n, K={0..7,16..23}; lane>=16: column n, K={8..15,24..31}.
static __device__ __forceinline__ v16bf load_bfrag(const bf16_t* colPtr, int hl) {
    v8bf lo = *(const v8bf*)(colPtr + hl * 8);
    v8bf hi = *(const v8bf*)(colPtr + 16 + hl * 8);
    return cat16(lo, hi);
}

// ---- CDNA5 async global->LDS copy (16B per lane), tracked by ASYNCcnt ----
static __device__ __forceinline__ void async_copy_b128(void* ldsDst, const void* gSrc) {
    uint32_t dst = (uint32_t)(uintptr_t)ldsDst;           // LDS byte offset
    uint64_t src = (uint64_t)(uintptr_t)gSrc;
    asm volatile("global_load_async_to_lds_b128 %0, %1, off"
                 :: "v"(dst), "v"(src) : "memory");
}
static __device__ __forceinline__ void wait_async0() {
    asm volatile("s_wait_asynccnt 0x0" ::: "memory");
}

// ---- CDNA5 Tensor Data Mover: 2D tile load, bf16 elements, LDS row padding ----
// tile_w elems per row (row byte size must be DWORD multiple), tile_h rows,
// global row stride in elems; LDS pad: +pad_amt_dw DWORDs after every 2^(pad_int+1) DWORDs.
static __device__ __forceinline__ void tdm_load_2d(uint32_t lds_off, const void* gaddr,
                                                   uint32_t tile_w, uint32_t tile_h,
                                                   uint32_t row_stride,
                                                   uint32_t pad_int_code,
                                                   uint32_t pad_amt_code) {
    uint64_t ga = (uint64_t)(uintptr_t)gaddr;
    u32x4 g0;
    g0[0] = 1u;                                            // count=1, user descriptor
    g0[1] = lds_off;                                       // lds_addr
    g0[2] = (uint32_t)ga;                                  // global_addr[31:0]
    g0[3] = (uint32_t)((ga >> 32) & 0x01FFFFFFu) | (2u << 30); // addr[56:32] | type=2
    u32x8 g1;
    g1[0] = (1u << 16)                                     // data_size = 1 (2 bytes)
          | (1u << 20)                                     // pad_enable
          | (pad_int_code << 22) | (pad_amt_code << 25);
    g1[1] = (tile_w & 0xFFFFu) << 16;                      // tensor_dim0[15:0] (barrier addr 0)
    g1[2] = ((tile_w >> 16) & 0xFFFFu) | ((tile_h & 0xFFFFu) << 16); // dim0 hi | dim1 lo
    g1[3] = ((tile_h >> 16) & 0xFFFFu) | (tile_w << 16);   // dim1 hi | tile_dim0
    g1[4] = tile_h & 0xFFFFu;                              // tile_dim1 (tile_dim2 = 0)
    g1[5] = row_stride;                                    // tensor_dim0_stride lo32
    g1[6] = 0u;                                            // stride hi | dim1_stride lo
    g1[7] = 0u;
    asm volatile("tensor_load_to_lds %0, %1" :: "s"(g0), "s"(g1) : "memory");
}

// ---------------- kernel 0: weight transpose + bf16 convert ----------------
__global__ __launch_bounds__(256)
void prep_weights_kernel(const float* __restrict__ qw, const float* __restrict__ ow,
                         bf16_t* __restrict__ wtq, bf16_t* __restrict__ wto) {
    int id = blockIdx.x * 256 + threadIdx.x;
    if (id < N3 * C) {
        int n = id >> 9, kk = id & (C - 1);
        wtq[id] = (bf16_t)qw[kk * N3 + n];
    } else {
        int id2 = id - N3 * C;
        int n = id2 >> 9, kk = id2 & (C - 1);
        wto[id2] = (bf16_t)ow[kk * C + n];
    }
}

// ---------------- kernel 1: QKV projection GEMM (double-buffered) ----------------
// x (f32) tile: global load + v_cvt_pk_bf16_f32 + ds_store into next buffer.
// weight tile: async global->LDS copy. Scatter epilogue into q,k,(padded) v.
__global__ __launch_bounds__(256)
void qkv_gemm_kernel(const float* __restrict__ x, const bf16_t* __restrict__ wt,
                     const float* __restrict__ qkv_b,
                     bf16_t* __restrict__ qo, bf16_t* __restrict__ ko,
                     bf16_t* __restrict__ vo) {
    __shared__ bf16_t As[2][64][40];
    __shared__ bf16_t Bs[2][64][40];

    const int n0 = blockIdx.x * 64;
    const int m0 = blockIdx.y * 64;
    const int tid  = threadIdx.x;
    const int wid  = tid >> 5, lane = tid & 31;
    const int rt   = wid >> 1, cg = wid & 1;
    const int hl   = lane >> 4, lrow = lane & 15;
    const int srow = tid >> 2, sseg = tid & 3;

    const float*  xrow = x  + (size_t)(m0 + srow) * C + sseg * 8;
    const bf16_t* wrow = wt + (size_t)(n0 + srow) * C + sseg * 8;

    auto stage = [&](int buf, int k0) {
        async_copy_b128(&Bs[buf][srow][sseg * 8], wrow + k0);
        v4f f0 = *(const v4f*)(xrow + k0);
        v4f f1 = *(const v4f*)(xrow + k0 + 4);
        v8bf a8;
        #pragma unroll
        for (int i = 0; i < 4; ++i) { a8[i] = (bf16_t)f0[i]; a8[4 + i] = (bf16_t)f1[i]; }
        *(v8bf*)(&As[buf][srow][sseg * 8]) = a8;
    };

    v8f c0 = zero8(), c1 = zero8();

    stage(0, 0);
    wait_async0();
    __syncthreads();

    for (int k0 = 0, it = 0; k0 < C; k0 += 32, ++it) {
        int cur = it & 1;
        if (k0 + 32 < C) {
            __builtin_prefetch(xrow + k0 + 64, 0, 3);
            stage(cur ^ 1, k0 + 32);
        }
        v16bf af = load_afrag(&As[cur][rt * 16 + lrow][0], hl);
        v16bf b0 = load_bfrag(&Bs[cur][cg * 32 + lrow][0], hl);
        v16bf b1 = load_bfrag(&Bs[cur][cg * 32 + 16 + lrow][0], hl);
        c0 = wmma_bf16(af, b0, c0);
        c1 = wmma_bf16(af, b1, c1);
        wait_async0();
        __syncthreads();
    }

    const float scale = 0.17677669529663687f; // 1/sqrt(32)
    #pragma unroll
    for (int ct = 0; ct < 2; ++ct) {
        v8f acc = ct ? c1 : c0;
        int n = n0 + cg * 32 + ct * 16 + lrow;
        int e = n >> 9;
        int h = (n >> 5) & (H - 1);
        int d = n & (D - 1);
        float bias = qkv_b[n];
        #pragma unroll
        for (int r = 0; r < 8; ++r) {
            int grow = m0 + rt * 16 + r + 8 * hl;
            int b = grow / S, s = grow - b * S;
            float val = acc[r] + bias;
            size_t head = (size_t)(b * H + h);
            if (e == 0)      qo[head * (S * D)  + s * D  + d] = (bf16_t)(val * scale);
            else if (e == 1) ko[head * (S * D)  + s * D  + d] = (bf16_t)val;
            else             vo[head * (D * SP) + d * SP + s] = (bf16_t)val; // d-major, padded
        }
    }
}

// ---------------- kernel 2: fused window attention per (b,h), TDM staging ----------------
__global__ __launch_bounds__(256)
void attn_kernel(const bf16_t* __restrict__ q, const bf16_t* __restrict__ k,
                 const bf16_t* __restrict__ v, const float* __restrict__ mask,
                 const float* __restrict__ bias_table, const int* __restrict__ rel_index,
                 bf16_t* __restrict__ o) {
    __shared__ bf16_t Qs[SP][40];   // padded tokens x d   (TDM: 64B rows + 16B pad)
    __shared__ bf16_t Ks[SP][40];
    __shared__ bf16_t Vt[D][72];    // d x padded tokens   (TDM: 128B rows + 16B pad)
    __shared__ float  Sc[SP][68];   // f32 scores
    __shared__ bf16_t At[SP][72];   // softmaxed probs, bf16

    const int bh = blockIdx.x;
    const int b  = bh >> 4, h = bh & (H - 1);
    const int w  = b & (NW - 1);
    const int tid = threadIdx.x;
    const int wid = tid >> 5, lane = tid & 31;
    const int hl  = lane >> 4, lrow = lane & 15;

    const bf16_t* qb = q + (size_t)bh * (S * D);
    const bf16_t* kb = k + (size_t)bh * (S * D);
    const bf16_t* vb = v + (size_t)bh * (D * SP);

    // ---- TDM loads (one wave issues; EXEC-independent, per-wave TENSORcnt) ----
    if (wid == 0) {
        // Q,K: 32-elem rows x 49, pad 16B after every 16 DWORDs -> stride 40 elems
        tdm_load_2d((uint32_t)(uintptr_t)&Qs[0][0], qb, D, S, D, 3u, 3u);
        tdm_load_2d((uint32_t)(uintptr_t)&Ks[0][0], kb, D, S, D, 3u, 3u);
        // Vt: 64-elem rows x 32, pad 16B after every 32 DWORDs -> stride 72 elems
        tdm_load_2d((uint32_t)(uintptr_t)&Vt[0][0], vb, SP, D, SP, 4u, 3u);
    }
    // ---- zero-fill Q/K pad rows (49..63) while TDM runs ----
    {
        uint4 z = {0u, 0u, 0u, 0u};
        if (tid < 75) {            // 15 rows * 5 chunks of 16B
            int row = S + tid / 5, chunk = tid % 5;
            *(uint4*)(&Qs[row][chunk * 8]) = z;
        } else if (tid < 150) {
            int t2 = tid - 75;
            int row = S + t2 / 5, chunk = t2 % 5;
            *(uint4*)(&Ks[row][chunk * 8]) = z;
        }
    }
    if (wid == 0) __builtin_amdgcn_s_wait_tensorcnt((short)0);
    __syncthreads();

    // ---- scores = (q*scale) @ k^T : one wmma per 16x16 tile (K = d = 32) ----
    {
        int rtile = wid >> 1, chalf = wid & 1;
        v16bf af = load_afrag(&Qs[rtile * 16 + lrow][0], hl);
        #pragma unroll
        for (int ci = 0; ci < 2; ++ci) {
            int ct = chalf * 2 + ci;
            v16bf bf = load_bfrag(&Ks[ct * 16 + lrow][0], hl);
            v8f acc = wmma_bf16(af, bf, zero8());
            #pragma unroll
            for (int r = 0; r < 8; ++r)
                Sc[rtile * 16 + r + 8 * hl][ct * 16 + lrow] = acc[r];
        }
    }
    __syncthreads();

    // ---- bias gather + mask + softmax (pad cols excluded == -inf) ----
    if (tid < SP) {
        int s = tid;
        if (s < S) {
            const float* mrow = mask + ((size_t)w * S + s) * S;
            const int*   rrow = rel_index + s * S;
            float mx = -3.0e38f;
            for (int t = 0; t < S; ++t) {
                float val = Sc[s][t] + bias_table[rrow[t] * H + h] + mrow[t];
                Sc[s][t] = val;
                mx = fmaxf(mx, val);
            }
            float sum = 0.f;
            for (int t = 0; t < S; ++t) {
                float e = __expf(Sc[s][t] - mx);
                Sc[s][t] = e;
                sum += e;
            }
            float inv = 1.f / sum;
            for (int t = 0; t < S; ++t)  At[s][t] = (bf16_t)(Sc[s][t] * inv);
            for (int t = S; t < SP; ++t) At[s][t] = (bf16_t)0.f;
        } else {
            for (int t = 0; t < SP; ++t) At[s][t] = (bf16_t)0.f;
        }
    }
    __syncthreads();

    // ---- out = attn @ v : K = 64 (2 wmma steps), write (b,s,h,d) bf16 ----
    {
        int rtile = wid >> 1, dt = wid & 1;
        v8f acc = zero8();
        #pragma unroll
        for (int ks = 0; ks < 2; ++ks) {
            v16bf af = load_afrag(&At[rtile * 16 + lrow][ks * 32], hl);
            v16bf bf = load_bfrag(&Vt[dt * 16 + lrow][ks * 32], hl);
            acc = wmma_bf16(af, bf, acc);
        }
        #pragma unroll
        for (int r = 0; r < 8; ++r) {
            int s = rtile * 16 + r + 8 * hl;
            if (s < S)
                o[((size_t)b * S + s) * C + h * D + dt * 16 + lrow] = (bf16_t)acc[r];
        }
    }
}

// ---------------- kernel 3: out projection (fully async staged, double-buffered) ----------------
__global__ __launch_bounds__(256)
void out_gemm_kernel(const bf16_t* __restrict__ a, const bf16_t* __restrict__ wt,
                     const float* __restrict__ out_b, float* __restrict__ out) {
    __shared__ bf16_t As[2][64][40];
    __shared__ bf16_t Bs[2][64][40];

    const int n0 = blockIdx.x * 64;
    const int m0 = blockIdx.y * 64;
    const int tid  = threadIdx.x;
    const int wid  = tid >> 5, lane = tid & 31;
    const int rt   = wid >> 1, cg = wid & 1;
    const int hl   = lane >> 4, lrow = lane & 15;
    const int srow = tid >> 2, sseg = tid & 3;

    const bf16_t* arow = a  + (size_t)(m0 + srow) * C + sseg * 8;
    const bf16_t* wrow = wt + (size_t)(n0 + srow) * C + sseg * 8;

    auto stage = [&](int buf, int k0) {
        async_copy_b128(&As[buf][srow][sseg * 8], arow + k0);
        async_copy_b128(&Bs[buf][srow][sseg * 8], wrow + k0);
    };

    v8f c0 = zero8(), c1 = zero8();

    stage(0, 0);
    wait_async0();
    __syncthreads();

    for (int k0 = 0, it = 0; k0 < C; k0 += 32, ++it) {
        int cur = it & 1;
        if (k0 + 32 < C) stage(cur ^ 1, k0 + 32);
        v16bf af = load_afrag(&As[cur][rt * 16 + lrow][0], hl);
        v16bf b0 = load_bfrag(&Bs[cur][cg * 32 + lrow][0], hl);
        v16bf b1 = load_bfrag(&Bs[cur][cg * 32 + 16 + lrow][0], hl);
        c0 = wmma_bf16(af, b0, c0);
        c1 = wmma_bf16(af, b1, c1);
        wait_async0();
        __syncthreads();
    }

    #pragma unroll
    for (int ct = 0; ct < 2; ++ct) {
        v8f acc = ct ? c1 : c0;
        int n = n0 + cg * 32 + ct * 16 + lrow;
        float bias = out_b[n];
        #pragma unroll
        for (int r = 0; r < 8; ++r) {
            int grow = m0 + rt * 16 + r + 8 * hl;
            out[(size_t)grow * C + n] = acc[r] + bias;
        }
    }
}

// ---------------- host launcher ----------------
extern "C" void kernel_launch(void* const* d_in, const int* in_sizes, int n_in,
                              void* d_out, int out_size, void* d_ws, size_t ws_size,
                              hipStream_t stream) {
    const float* x          = (const float*)d_in[0];
    const float* mask       = (const float*)d_in[1];
    const float* qkv_w      = (const float*)d_in[2];
    const float* qkv_b      = (const float*)d_in[3];
    const float* out_w      = (const float*)d_in[4];
    const float* out_b      = (const float*)d_in[5];
    const float* bias_table = (const float*)d_in[6];
    const int*   rel_index  = (const int*)d_in[7];

    const size_t qk_sz  = (size_t)WB * H * S * D * sizeof(bf16_t);  // 102,760,448
    const size_t v_sz   = (size_t)WB * H * D * SP * sizeof(bf16_t); // 134,217,728 (padded)
    const size_t ao_sz  = (size_t)WB * S * C * sizeof(bf16_t);      // 102,760,448
    const size_t wtq_sz = (size_t)N3 * C * sizeof(bf16_t);
    const size_t wto_sz = (size_t)C * C * sizeof(bf16_t);
    const size_t need   = 2 * qk_sz + v_sz + ao_sz + wtq_sz + wto_sz;
    if (ws_size < need) return;

    char* ws = (char*)d_ws;
    bf16_t* q   = (bf16_t*)ws;  ws += qk_sz;
    bf16_t* k   = (bf16_t*)ws;  ws += qk_sz;
    bf16_t* v   = (bf16_t*)ws;  ws += v_sz;
    bf16_t* ao  = (bf16_t*)ws;  ws += ao_sz;
    bf16_t* wtq = (bf16_t*)ws;  ws += wtq_sz;
    bf16_t* wto = (bf16_t*)ws;  ws += wto_sz;

    // zero V so its pad columns (t = 49..63) contribute exact zeros in attn@V
    hipMemsetAsync(v, 0, v_sz, stream);

    prep_weights_kernel<<<(N3 * C + C * C) / 256, 256, 0, stream>>>(qkv_w, out_w, wtq, wto);
    qkv_gemm_kernel<<<dim3(N3 / 64, MROWS / 64), 256, 0, stream>>>(x, wtq, qkv_b, q, k, v);
    attn_kernel<<<WB * H, 256, 0, stream>>>(q, k, v, mask, bias_table, rel_index, ao);
    out_gemm_kernel<<<dim3(C / 64, MROWS / 64), 256, 0, stream>>>(ao, wto, out_b, (float*)d_out);
}
